// Attention_41841571398077
// MI455X (gfx1250) — matrix operand order repaired and checked by reading, not compiled, not run on hardware
//
#include <hip/hip_runtime.h>
#include <hip/hip_bf16.h>

// Problem dims
#define BB   256
#define AA   196
#define DV   2048
#define RNN  1024
#define HH   512
#define KCH  64

typedef __attribute__((ext_vector_type(16))) __bf16 v16bf;
typedef __attribute__((ext_vector_type(8)))  __bf16 v8bf;
typedef __attribute__((ext_vector_type(8)))  float  v8f;
typedef __attribute__((ext_vector_type(4)))  unsigned int u32x4;
typedef __attribute__((ext_vector_type(8)))  int   i32x8;
typedef __attribute__((ext_vector_type(4)))  int   i32x4;

typedef __attribute__((address_space(3))) void lds_void;

static __device__ __forceinline__ unsigned ldsOffsetOf(void* p) {
    lds_void* q = (lds_void*)p;                 // addrspacecast generic -> LDS
    return (unsigned)(unsigned long long)q;     // 32-bit LDS byte offset
}

// Issue one TDM 2D tile load: Wvt[kc .. ][512 cols] -> padded LDS rows of 72 bf16
static __device__ __forceinline__ void tdm_load_B(unsigned ldsOff,
                                                  const unsigned short* gsrc)
{
    unsigned long long ga = (unsigned long long)(size_t)gsrc;
    u32x4 g0;
    g0[0] = 1u;                                          // count=1, no gather
    g0[1] = ldsOff;                                      // lds_addr
    g0[2] = (unsigned)(ga & 0xFFFFFFFFu);                // global_addr[31:0]
    g0[3] = (unsigned)((ga >> 32) & 0x1FFFFFFu)          // global_addr[56:32]
          | 0x80000000u;                                 // type=2 ("image")
    i32x8 g1;
    g1[0] = (int)((1u << 16)      // data_size = 2 bytes
                | (1u << 20)      // pad_enable
                | (4u << 22)      // pad_interval: 32 DWORDs (=128B tile row)
                | (3u << 25));    // pad_amount: 4 DWORDs (=16B row pad)
    g1[1] = (int)((unsigned)(DV  & 0xFFFF) << 16);   // tensor_dim0 = 2048 (lo16)
    g1[2] = (int)((unsigned)(HH  & 0xFFFF) << 16);   // dim0 hi=0 | tensor_dim1 = 512
    g1[3] = (int)((unsigned)(KCH & 0xFFFF) << 16);   // dim1 hi=0 | tile_dim0 = 64
    g1[4] = (int)(HH);                               // tile_dim1 = 512, tile_dim2 = 0
    g1[5] = (int)(DV);                               // tensor_dim0_stride = 2048 (lo32)
    g1[6] = 0;                                       // stride hi16 | dim1_stride lo16
    g1[7] = 0;
    i32x4 z4 = {0, 0, 0, 0};
#if defined(__clang_major__) && (__clang_major__ >= 23)
    i32x8 z8 = {0, 0, 0, 0, 0, 0, 0, 0};
    __builtin_amdgcn_tensor_load_to_lds(g0, g1, z4, z4, z8, 0);
#else
    __builtin_amdgcn_tensor_load_to_lds(g0, g1, z4, z4, 0);
#endif
}

// ---------------------------------------------------------------------------
// Kernel 1: c[b][h] = h_att[b,:]@W_ha[:,h] + prev_h2[b,:]@W_hv[:,h]
//                     + b_ha[h] + b_hv[h] + b_v[h]
// ---------------------------------------------------------------------------
__global__ __launch_bounds__(256) void hidden_proj_kernel(
    const float* __restrict__ h_att, const float* __restrict__ prev_h2,
    const float* __restrict__ W_ha,  const float* __restrict__ b_ha,
    const float* __restrict__ W_hv,  const float* __restrict__ b_hv,
    const float* __restrict__ b_v,   float* __restrict__ c_ws)
{
    __shared__ float ha[RNN];
    __shared__ float hv[RNN];
    const int b   = blockIdx.x;
    const int tid = threadIdx.x;
    for (int k = tid; k < RNN; k += 256) {
        ha[k] = h_att[b * RNN + k];
        hv[k] = prev_h2[b * RNN + k];
    }
    __syncthreads();
    for (int h = tid; h < HH; h += 256) {
        float acc = b_ha[h] + b_hv[h] + b_v[h];
        #pragma unroll 4
        for (int k = 0; k < RNN; ++k) {
            acc = fmaf(ha[k], W_ha[k * HH + h], acc);
            acc = fmaf(hv[k], W_hv[k * HH + h], acc);
        }
        c_ws[b * HH + h] = acc;
    }
}

// ---------------------------------------------------------------------------
// Kernel 2: W_v [DV, HH] f32  ->  Wvt [HH, DV] bf16 (transposed, K-contiguous)
// ---------------------------------------------------------------------------
__global__ __launch_bounds__(256) void convert_wv_kernel(
    const float* __restrict__ W_v, unsigned short* __restrict__ Wvt_raw)
{
    __bf16* Wvt = (__bf16*)Wvt_raw;
    int idx = blockIdx.x * 256 + threadIdx.x;          // 0 .. HH*DV-1
    int h = idx >> 11;                                  // / DV
    int k = idx & (DV - 1);
    Wvt[idx] = (__bf16)W_v[k * HH + h];
}

// ---------------------------------------------------------------------------
// Kernel 3: fused  att[b,a] = relu(imgs[b,a,:]@W_v + c[b,:]) @ W_f + b_f
// 784 blocks x 64 rows of flattened (b,a).
// 512 threads = 16 waves: 4 row-subtiles x 4 column-quarters (128 cols each)
// -> only 8 v8f accumulators/wave (64 VGPRs), so the compiler has headroom to
//    double-buffer the B-fragment LDS loads across WMMAs.
// Double-buffered LDS; B tile staged by the Tensor Data Mover.
// ---------------------------------------------------------------------------
__global__ __launch_bounds__(512) void att_gemm_kernel(
    const float*          __restrict__ imgs,
    const unsigned short* __restrict__ Wvt_raw,
    const float*          __restrict__ c_ws,
    const float*          __restrict__ W_f,
    const float*          __restrict__ b_f,
    float*                __restrict__ att_ws)
{
    __shared__ __align__(16) __bf16 As[2][64][72];   // 18.4 KB
    __shared__ __align__(16) __bf16 Bs[2][HH][72];   // 147.5 KB
    __shared__ float att_sh[64];

    const int tid  = threadIdx.x;
    const int lane = tid & 31;
    const int wave = tid >> 5;          // 0..15
    const int half = (lane >> 4) & 1;   // which half-wave
    const int m16  = lane & 15;
    const int rs   = wave & 3;          // row subtile 0..3
    const int cq   = wave >> 2;         // column quarter 0..3 (128 cols each)
    const int rowTileBase = blockIdx.x * 64;

    if (tid < 64) att_sh[tid] = 0.0f;

    v8f acc[8];
    #pragma unroll
    for (int t = 0; t < 8; ++t)
        #pragma unroll
        for (int i = 0; i < 8; ++i) acc[t][i] = 0.0f;

    const int rowA = rs * 16 + m16;     // local row this lane's A-frag comes from

    // per-thread A staging assignment: 64 rows x 64 K f32, 8 floats each
    const int rl = tid >> 3;            // 0..63
    const int k0 = (tid & 7) << 3;      // 0,8,...,56
    const long long rowStageBase = (long long)(rowTileBase + rl) * DV + k0;

    float4 fa[2];

    // ---- prologue: stage chunk 0 into buffer 0 ----
    {
        const float* src = imgs + rowStageBase;
        fa[0] = ((const float4*)src)[0];
        fa[1] = ((const float4*)src)[1];
        if (wave == 0)
            tdm_load_B(ldsOffsetOf(&Bs[0][0][0]), Wvt_raw);
        #pragma unroll
        for (int j = 0; j < 2; ++j) {
            As[0][rl][k0 + 4 * j + 0] = (__bf16)fa[j].x;
            As[0][rl][k0 + 4 * j + 1] = (__bf16)fa[j].y;
            As[0][rl][k0 + 4 * j + 2] = (__bf16)fa[j].z;
            As[0][rl][k0 + 4 * j + 3] = (__bf16)fa[j].w;
        }
        if (wave == 0) __builtin_amdgcn_s_wait_tensorcnt(0);
        __syncthreads();
    }

    for (int kc = 0; kc < DV; kc += KCH) {
        const int  cur  = (kc / KCH) & 1;
        const int  nxt  = cur ^ 1;
        const bool more = (kc + KCH) < DV;

        // ---- issue next chunk's A loads + TDM B load (overlap with compute) ----
        if (more) {
            const float* src = imgs + rowStageBase + kc + KCH;
            __builtin_prefetch(src + KCH, 0, 0);       // global_prefetch_b8
            fa[0] = ((const float4*)src)[0];
            fa[1] = ((const float4*)src)[1];
            if (wave == 0)
                tdm_load_B(ldsOffsetOf(&Bs[nxt][0][0]), Wvt_raw + kc + KCH);
        }

        // ---- compute on current buffers: 2 K-steps x 8 column tiles ----
        #pragma unroll
        for (int kk = 0; kk < KCH; kk += 32) {
            // A frag per ISA layout: lanes0-15 K{0..7,16..23}, lanes16-31 K{8..15,24..31}
            v8bf alo = *(const v8bf*)&As[cur][rowA][kk + half * 8];
            v8bf ahi = *(const v8bf*)&As[cur][rowA][kk + 16 + half * 8];
            v16bf afrag = __builtin_shufflevector(alo, ahi,
                0, 1, 2, 3, 4, 5, 6, 7, 8, 9, 10, 11, 12, 13, 14, 15);

            const int col0 = (cq << 7) + m16;
            // software-pipeline the B-fragment LDS loads one tile ahead
            v8bf blo = *(const v8bf*)&Bs[cur][col0][kk + half * 8];
            v8bf bhi = *(const v8bf*)&Bs[cur][col0][kk + 16 + half * 8];
            #pragma unroll
            for (int t = 0; t < 8; ++t) {
                v8bf nlo = blo, nhi = bhi;
                if (t < 7) {
                    const int coln = col0 + ((t + 1) << 4);
                    nlo = *(const v8bf*)&Bs[cur][coln][kk + half * 8];
                    nhi = *(const v8bf*)&Bs[cur][coln][kk + 16 + half * 8];
                }
                v16bf bfrag = __builtin_shufflevector(blo, bhi,
                    0, 1, 2, 3, 4, 5, 6, 7, 8, 9, 10, 11, 12, 13, 14, 15);
                acc[t] = __builtin_amdgcn_wmma_f32_16x16x32_bf16(
                    false, afrag, false, bfrag, (short)0, acc[t], false, false);
                blo = nlo; bhi = nhi;
            }
        }

        // ---- finish next chunk's staging, then publish ----
        if (more) {
            #pragma unroll
            for (int j = 0; j < 2; ++j) {
                As[nxt][rl][k0 + 4 * j + 0] = (__bf16)fa[j].x;
                As[nxt][rl][k0 + 4 * j + 1] = (__bf16)fa[j].y;
                As[nxt][rl][k0 + 4 * j + 2] = (__bf16)fa[j].z;
                As[nxt][rl][k0 + 4 * j + 3] = (__bf16)fa[j].w;
            }
            if (wave == 0) __builtin_amdgcn_s_wait_tensorcnt(0);
        }
        __syncthreads();
    }

    // ---- epilogue: +c[b,h], relu, *W_f[h], reduce over N ----
    float s[8];
    #pragma unroll
    for (int r = 0; r < 8; ++r) s[r] = 0.0f;

    #pragma unroll
    for (int t = 0; t < 8; ++t) {
        const int h  = (cq << 7) + (t << 4) + m16;   // N column of this lane
        const float wf = W_f[h];
        #pragma unroll
        for (int r = 0; r < 8; ++r) {
            const int row = rowTileBase + rs * 16 + r + half * 8;  // D: M=r+8*half
            const int b   = row / AA;
            float v = acc[t][r] + c_ws[b * HH + h];
            v = fmaxf(v, 0.0f);
            s[r] = fmaf(v, wf, s[r]);
        }
    }
    // reduce across the 16 N-lanes of each half-wave
    #pragma unroll
    for (int r = 0; r < 8; ++r) {
        #pragma unroll
        for (int off = 1; off < 16; off <<= 1)
            s[r] += __shfl_xor(s[r], off, 32);
    }
    if (m16 == 0) {
        #pragma unroll
        for (int r = 0; r < 8; ++r)
            atomicAdd(&att_sh[rs * 16 + r + half * 8], s[r]);
    }
    __syncthreads();

    if (tid < 64) {
        const int row = rowTileBase + tid;
        const int b = row / AA;
        const int a = row - b * AA;
        att_ws[b * AA + a] = att_sh[tid] + b_f[0];
    }
}

// ---------------------------------------------------------------------------
// Kernel 4: softmax over A, then out[b,:] = sum_a alpha[b,a] * imgs[b,a,:]
// ---------------------------------------------------------------------------
__global__ __launch_bounds__(256) void softmax_wsum_kernel(
    const float* __restrict__ imgs, const float* __restrict__ att_ws,
    float* __restrict__ out)
{
    __shared__ float sdata[256];
    __shared__ float alpha[256];
    const int b   = blockIdx.x;
    const int tid = threadIdx.x;

    float v = (tid < AA) ? att_ws[b * AA + tid] : -3.4e38f;
    sdata[tid] = v;
    __syncthreads();
    for (int s = 128; s > 0; s >>= 1) {
        if (tid < s) sdata[tid] = fmaxf(sdata[tid], sdata[tid + s]);
        __syncthreads();
    }
    const float mx = sdata[0];
    __syncthreads();
    float e = (tid < AA) ? __expf(v - mx) : 0.0f;
    sdata[tid] = e;
    __syncthreads();
    for (int s = 128; s > 0; s >>= 1) {
        if (tid < s) sdata[tid] += sdata[tid + s];
        __syncthreads();
    }
    alpha[tid] = e * (1.0f / sdata[0]);
    __syncthreads();

    const float* base = imgs + (long long)b * AA * DV;
    for (int d = tid; d < DV; d += 256) {
        float accv = 0.0f;
        for (int a = 0; a < AA; ++a)
            accv = fmaf(alpha[a], base[(long long)a * DV + d], accv);
        out[(long long)b * DV + d] = accv;
    }
}

// ---------------------------------------------------------------------------
extern "C" void kernel_launch(void* const* d_in, const int* in_sizes, int n_in,
                              void* d_out, int out_size, void* d_ws, size_t ws_size,
                              hipStream_t stream)
{
    const float* h_att   = (const float*)d_in[0];
    const float* prev_h2 = (const float*)d_in[1];
    const float* imgs    = (const float*)d_in[2];
    const float* W_v     = (const float*)d_in[3];
    const float* b_v     = (const float*)d_in[4];
    const float* W_ha    = (const float*)d_in[5];
    const float* b_ha    = (const float*)d_in[6];
    const float* W_hv    = (const float*)d_in[7];
    const float* b_hv    = (const float*)d_in[8];
    const float* W_f     = (const float*)d_in[9];
    const float* b_f     = (const float*)d_in[10];
    float* out = (float*)d_out;

    // workspace layout: c[B,H] f32 | Wvt[H,DV] bf16 | att[B,A] f32  (~2.7 MB)
    char* ws = (char*)d_ws;
    float*          c_ws   = (float*)ws;                              // 512 KB
    unsigned short* Wvt    = (unsigned short*)(ws + (size_t)BB * HH * 4);
    float*          att_ws = (float*)(ws + (size_t)BB * HH * 4
                                         + (size_t)HH * DV * 2);      // 200 KB

    hidden_proj_kernel<<<BB, 256, 0, stream>>>(h_att, prev_h2, W_ha, b_ha,
                                               W_hv, b_hv, b_v, c_ws);
    convert_wv_kernel<<<(HH * DV) / 256, 256, 0, stream>>>(W_v, Wvt);
    att_gemm_kernel<<<(BB * AA) / 64, 512, 0, stream>>>(imgs, Wvt, c_ws,
                                                        W_f, b_f, att_ws);
    softmax_wsum_kernel<<<BB, 256, 0, stream>>>(imgs, att_ws, out);
}
